// GlmImageVQVAEVectorQuantizer_15384572854609
// MI455X (gfx1250) — matrix operand order repaired and checked by reading, not compiled, not run on hardware
//
#include <hip/hip_runtime.h>
#include <hip/hip_bf16.h>

typedef __attribute__((ext_vector_type(16))) __bf16 v16bf;
typedef __attribute__((ext_vector_type(8)))  float  v8f;

#define D_DIM   2048
#define N_ROWS  8192     // B*H*W
#define K_EMB   16384
#define HW_     1024

// ---------- helpers ----------
static __device__ inline unsigned short f2bf(float f) {
  unsigned u = __builtin_bit_cast(unsigned, f);
  unsigned r = u + 0x7FFFu + ((u >> 16) & 1u);   // round-to-nearest-even
  return (unsigned short)(r >> 16);
}

union ABFrag {            // 32 bytes = 16 bf16 = 8 VGPRs per lane
  uint4 q[2];
  v16bf v;
};

// ---------- kernel 1: L2-normalize hidden (NCHW -> [N,D]) to bf16 ----------
__global__ __launch_bounds__(256) void k_norm_hidden(
    const float* __restrict__ hs, unsigned short* __restrict__ zb) {
  int n    = blockIdx.x * 8 + (threadIdx.x >> 5);   // one wave per row
  int lane = threadIdx.x & 31;
  int b = n >> 10, hw = n & 1023;
  const float* src = hs + (size_t)b * D_DIM * HW_ + hw;
  float vals[64];
  float ss = 0.f;
#pragma unroll
  for (int i = 0; i < 64; ++i) {
    int c = lane + 32 * i;
    float v = src[(size_t)c * HW_];
    vals[i] = v;
    ss += v * v;
  }
#pragma unroll
  for (int m = 16; m >= 1; m >>= 1) ss += __shfl_xor(ss, m, 32);
  float inv = 1.f / fmaxf(sqrtf(ss), 1e-12f);
  unsigned short* dst = zb + (size_t)n * D_DIM;
#pragma unroll
  for (int i = 0; i < 64; ++i) dst[lane + 32 * i] = f2bf(vals[i] * inv);
}

// ---------- kernel 2: L2-normalize codebook to bf16 + store inv norm ----------
__global__ __launch_bounds__(256) void k_norm_embed(
    const float* __restrict__ em, unsigned short* __restrict__ eb,
    float* __restrict__ inv_e) {
  int k    = blockIdx.x * 8 + (threadIdx.x >> 5);
  int lane = threadIdx.x & 31;
  const float4* src = (const float4*)(em + (size_t)k * D_DIM);
  float4 vals[16];
  float ss = 0.f;
#pragma unroll
  for (int i = 0; i < 16; ++i) {
    float4 v = src[lane + 32 * i];
    vals[i] = v;
    ss += v.x * v.x + v.y * v.y + v.z * v.z + v.w * v.w;
  }
#pragma unroll
  for (int m = 16; m >= 1; m >>= 1) ss += __shfl_xor(ss, m, 32);
  float inv = 1.f / fmaxf(sqrtf(ss), 1e-12f);
  if (lane == 0) inv_e[k] = inv;
  ushort4* dst = (ushort4*)(eb + (size_t)k * D_DIM);
#pragma unroll
  for (int i = 0; i < 16; ++i) {
    float4 v = vals[i];
    ushort4 o;
    o.x = f2bf(v.x * inv); o.y = f2bf(v.y * inv);
    o.z = f2bf(v.z * inv); o.w = f2bf(v.w * inv);
    dst[lane + 32 * i] = o;
  }
}

// ---------- kernel 3: WMMA GEMM (z * e^T) + running argmax ----------
// grid (32 row-blocks of 256 rows, 8 col-splits of 2048 cols); block 256 = 8 waves.
// Wave tile: 32 rows x 128 cols = 2 A-frags x 8 B-frags -> 16 WMMA / k-step.
// LDS rows padded to 272B so 16-lane ds_load_b128 hits banks 0,4,...,60.
#define KC       128
#define LROWB    272
#define SA_BYTES (256 * LROWB)   // 69,632
#define SB_BYTES (128 * LROWB)   // 34,816
__global__ __launch_bounds__(256) void k_gemm_argmax(
    const unsigned short* __restrict__ zb, const unsigned short* __restrict__ eb,
    float* __restrict__ pval, int* __restrict__ pidx) {
  extern __shared__ char smem[];
  char* sA = smem;                // 256 rows x 128 bf16 (padded)
  char* sB = smem + SA_BYTES;     // 128 cols x 128 bf16 (padded)

  int tid  = threadIdx.x;
  int wave = tid >> 5, lane = tid & 31;
  int half = lane >> 4, l15 = lane & 15;
  int rowbase = blockIdx.x * 256;
  int s       = blockIdx.y;            // col split: [s*2048, s*2048+2048)

  float bestv[2][8];
  int   besti[2][8];
#pragma unroll
  for (int rr = 0; rr < 2; ++rr)
#pragma unroll
    for (int v = 0; v < 8; ++v) { bestv[rr][v] = -3.4e38f; besti[rr][v] = 0; }

  const v8f vzero = {0.f, 0.f, 0.f, 0.f, 0.f, 0.f, 0.f, 0.f};

  for (int cb = 0; cb < 16; ++cb) {
    int colbase = s * 2048 + cb * 128;
    v8f acc[2][8];
#pragma unroll
    for (int rr = 0; rr < 2; ++rr)
#pragma unroll
      for (int t = 0; t < 8; ++t) acc[rr][t] = vzero;

    for (int kc = 0; kc < D_DIM; kc += KC) {
      __syncthreads();                 // previous LDS consumers done
      // stage A: 256x128 bf16 = 4096 uint4; B: 128x128 bf16 = 2048 uint4
#pragma unroll
      for (int i = 0; i < 16; ++i) {
        int u = tid + 256 * i;
        int row = u >> 4, c16 = u & 15;
        const char* ga = (const char*)zb +
            ((size_t)(rowbase + row) * D_DIM + kc) * 2 + c16 * 16;
        *(uint4*)(sA + row * LROWB + c16 * 16) = *(const uint4*)ga;
      }
#pragma unroll
      for (int i = 0; i < 8; ++i) {
        int u = tid + 256 * i;
        int row = u >> 4, c16 = u & 15;
        const char* gb = (const char*)eb +
            ((size_t)(colbase + row) * D_DIM + kc) * 2 + c16 * 16;
        *(uint4*)(sB + row * LROWB + c16 * 16) = *(const uint4*)gb;
      }
      __syncthreads();

#pragma unroll
      for (int kk = 0; kk < KC; kk += 32) {
        // Two A fragments: rows wave*32 + {l15, 16+l15}
        // lanes<16 take k[0..7]+k[16..23], lanes>=16 k[8..15]+k[24..31]
        ABFrag af[2];
#pragma unroll
        for (int rr = 0; rr < 2; ++rr) {
          const char* ap = sA + (wave * 32 + rr * 16 + l15) * LROWB
                              + (kk + half * 8) * 2;
          af[rr].q[0] = *(const uint4*)ap;
          af[rr].q[1] = *(const uint4*)(ap + 32);
        }
#pragma unroll
        for (int t = 0; t < 8; ++t) {
          // B fragment: lanes<16 take k[0..7]+k[16..23], lanes>=16 k[8..15]+k[24..31]
          ABFrag bfr;
          const char* bp = sB + (t * 16 + l15) * LROWB + (kk + half * 8) * 2;
          bfr.q[0] = *(const uint4*)bp;
          bfr.q[1] = *(const uint4*)(bp + 32);
          acc[0][t] = __builtin_amdgcn_wmma_f32_16x16x32_bf16(
              false, af[0].v, false, bfr.v, (short)0, acc[0][t], false, false);
          acc[1][t] = __builtin_amdgcn_wmma_f32_16x16x32_bf16(
              false, af[1].v, false, bfr.v, (short)0, acc[1][t], false, false);
        }
      }
    }

    // fold this 128-col block into per-lane running argmax
#pragma unroll
    for (int t = 0; t < 8; ++t) {
      int ng = colbase + t * 16 + l15;
#pragma unroll
      for (int rr = 0; rr < 2; ++rr)
#pragma unroll
        for (int v = 0; v < 8; ++v) {
          float f = acc[rr][t][v];
          if (f > bestv[rr][v]) { bestv[rr][v] = f; besti[rr][v] = ng; }
        }
    }
  }

  // reduce across the 16 lanes of each half (xor masks stay within half)
#pragma unroll
  for (int m = 8; m >= 1; m >>= 1) {
#pragma unroll
    for (int rr = 0; rr < 2; ++rr)
#pragma unroll
      for (int v = 0; v < 8; ++v) {
        float ov = __shfl_xor(bestv[rr][v], m, 32);
        int   oi = __shfl_xor(besti[rr][v], m, 32);
        if (ov > bestv[rr][v] || (ov == bestv[rr][v] && oi < besti[rr][v])) {
          bestv[rr][v] = ov; besti[rr][v] = oi;
        }
      }
  }
  if (l15 == 0) {
#pragma unroll
    for (int rr = 0; rr < 2; ++rr)
#pragma unroll
      for (int v = 0; v < 8; ++v) {
        int r = rowbase + wave * 32 + rr * 16 + half * 8 + v;
        pval[s * N_ROWS + r] = bestv[rr][v];
        pidx[s * N_ROWS + r] = besti[rr][v];
      }
  }
}

// ---------- kernel 3b: combine 8 col-split partials ----------
__global__ void k_argmax_combine(const float* __restrict__ pval,
                                 const int* __restrict__ pidx,
                                 int* __restrict__ idxo,
                                 float* __restrict__ out_idx_f) {
  int r = blockIdx.x * 256 + threadIdx.x;
  float best = pval[r];
  int   bi   = pidx[r];
#pragma unroll
  for (int s = 1; s < 8; ++s) {
    float v = pval[s * N_ROWS + r];
    int   i = pidx[s * N_ROWS + r];
    if (v > best || (v == best && i < bi)) { best = v; bi = i; }
  }
  idxo[r] = bi;
  out_idx_f[r] = (float)bi;   // idx output section (float dtype buffer)
}

// ---------- kernel 4: gather quant = e_norm[idx] in fp32, NHWC->NCHW ----------
__global__ __launch_bounds__(256) void k_gather(
    const float* __restrict__ em, const float* __restrict__ inv_e,
    const int* __restrict__ idxo, float* __restrict__ outq) {
  int c = blockIdx.x;   // 0..2047
  int b = blockIdx.y;   // 0..7
#pragma unroll
  for (int j = 0; j < 4; ++j) {
    int hw = threadIdx.x + 256 * j;
    int n  = b * HW_ + hw;
    int id = idxo[n];
    float v = em[(size_t)id * D_DIM + c] * inv_e[id];
    outq[((size_t)b * D_DIM + c) * HW_ + hw] = v;   // coalesced over hw
  }
}

// ---------- launcher ----------
extern "C" void kernel_launch(void* const* d_in, const int* in_sizes, int n_in,
                              void* d_out, int out_size, void* d_ws, size_t ws_size,
                              hipStream_t stream) {
  (void)in_sizes; (void)n_in; (void)out_size; (void)ws_size;
  const float* hidden = (const float*)d_in[0];   // [8,2048,32,32]
  const float* embed  = (const float*)d_in[1];   // [16384,2048]
  float* outq = (float*)d_out;                   // 16,777,216 quant floats
  float* outi = outq + 16777216;                 // then 8192 idx (as float)

  char* ws = (char*)d_ws;
  unsigned short* zb   = (unsigned short*)(ws + 0);            // 32 MB
  unsigned short* eb   = (unsigned short*)(ws + 33554432);     // 64 MB
  float*          inve = (float*)(ws + 100663296);             // 64 KB
  float*          pval = (float*)(ws + 100728832);             // 256 KB [8][8192]
  int*            pidx = (int*)  (ws + 100990976);             // 256 KB [8][8192]
  int*            idxf = (int*)  (ws + 101253120);             // 32 KB

  k_norm_hidden<<<N_ROWS / 8, 256, 0, stream>>>(hidden, zb);
  k_norm_embed <<<K_EMB  / 8, 256, 0, stream>>>(embed, eb, inve);

  dim3 g3(N_ROWS / 256, 8, 1);
  size_t lds = (size_t)SA_BYTES + SB_BYTES;      // 104,448 bytes
  k_gemm_argmax<<<g3, 256, lds, stream>>>(zb, eb, pval, pidx);

  k_argmax_combine<<<N_ROWS / 256, 256, 0, stream>>>(pval, pidx, idxf, outi);

  dim3 g4(D_DIM, 8, 1);
  k_gather<<<g4, 256, 0, stream>>>(embed, inve, idxf, outq);
}
